// HyperbolicRelCNN_62689342652473
// MI455X (gfx1250) — compile-verified, matrix operand order checked by reading, not run on hardware
//
#include <hip/hip_runtime.h>

#define EPSC  1e-07f
#define MINNC 1e-15f
#define MAXNC 1000000.0f

typedef __attribute__((ext_vector_type(16))) __bf16 v16bf;
typedef __attribute__((ext_vector_type(8)))  __bf16 v8bf;
typedef __attribute__((ext_vector_type(8)))  float  v8f;
typedef __attribute__((ext_vector_type(4)))  unsigned int u32x4;
typedef __attribute__((ext_vector_type(8)))  int i32x8;
typedef __attribute__((ext_vector_type(4)))  int i32x4;

#if defined(__has_builtin)
#  if __has_builtin(__builtin_amdgcn_tensor_load_to_lds) && __has_builtin(__builtin_amdgcn_s_wait_tensorcnt)
#    define HAVE_TDM 1
#  endif
#endif

// LDS layout for the staged B (weight) matrix:
//   slab s (128 k-columns) -> rows of 128+8 ushorts (272B stride => conflict-free
//   ds_load_b128 across 16 lanes on the 64-bank LDS).
#define BROW 136                 // ushorts per LDS row (128 data + 8 pad)
#define BSLAB (128 * BROW)       // ushorts per 128-column slab

// ---------- helpers ----------

__device__ __forceinline__ float wred32(float v) {
#pragma unroll
  for (int off = 16; off; off >>= 1) v += __shfl_xor(v, off, 32);
  return v;
}

__device__ __forceinline__ float arcosh_f(float x) {
  float xc = fmaxf(x, 1.0f);
  return logf(fmaxf(xc + sqrtf(fmaxf(xc * xc - 1.0f, 0.0f)), MINNC));
}

__device__ __forceinline__ unsigned short f2bf(float f) {
  unsigned int u = __builtin_bit_cast(unsigned int, f);
  unsigned int r = (u + 0x7FFFu + ((u >> 16) & 1u)) >> 16;  // RNE
  return (unsigned short)r;
}

// ---------- elementwise / conversion kernels ----------

__global__ void zero_f32_kernel(float* __restrict__ p, size_t n) {
  size_t i = (size_t)blockIdx.x * blockDim.x + threadIdx.x;
  size_t stride = (size_t)gridDim.x * blockDim.x;
  for (; i < n; i += stride) p[i] = 0.0f;
}

__global__ void f2bf_kernel(const float* __restrict__ s, unsigned short* __restrict__ d, int n) {
  int i = blockIdx.x * blockDim.x + threadIdx.x;
  if (i < n) d[i] = f2bf(s[i]);
}

__global__ void x_to_hcat_kernel(const float* __restrict__ x, unsigned short* __restrict__ hc, int N) {
  int i = blockIdx.x * blockDim.x + threadIdx.x;
  if (i < N * 128) {
    int node = i >> 7, k = i & 127;
    hc[(size_t)node * 384 + k] = f2bf(x[i]);
  }
}

// ---------- per-node: logmap0 -> bf16 (one wave per node, 4 f32/lane) ----------

__global__ void logmap0_bf16_kernel(const float* __restrict__ X, unsigned short* __restrict__ Y, int N) {
  int node = blockIdx.x * (blockDim.x >> 5) + (threadIdx.x >> 5);
  int lane = threadIdx.x & 31;
  if (node >= N) return;
  float4 xv = reinterpret_cast<const float4*>(X + (size_t)node * 128)[lane];
  float part = xv.x * xv.x + xv.y * xv.y + xv.z * xv.z + xv.w * xv.w;
  if (lane == 0) part -= xv.x * xv.x;                 // spatial norm only (k >= 1)
  float nsq = wred32(part);
  float yn  = fmaxf(sqrtf(nsq), MINNC);
  float x0  = __shfl(xv.x, 0, 32);
  float s   = arcosh_f(fmaxf(x0, 1.0f + EPSC)) / yn;
  float4 o;
  o.x = (lane == 0) ? 0.0f : s * xv.x;
  o.y = s * xv.y; o.z = s * xv.z; o.w = s * xv.w;
  union { unsigned short u[4]; uint2 v; } pk;
  pk.u[0] = f2bf(o.x); pk.u[1] = f2bf(o.y); pk.u[2] = f2bf(o.z); pk.u[3] = f2bf(o.w);
  reinterpret_cast<uint2*>(Y + (size_t)node * 128)[lane] = pk.v;
}

// ---------- per-node: expmap0 in-place on GEMM output ----------

__global__ void expmap0_kernel(float* __restrict__ T, int N) {
  int node = blockIdx.x * (blockDim.x >> 5) + (threadIdx.x >> 5);
  int lane = threadIdx.x & 31;
  if (node >= N) return;
  float4* row = reinterpret_cast<float4*>(T + (size_t)node * 128);
  float4 t = row[lane];
  float part = t.x * t.x + t.y * t.y + t.z * t.z + t.w * t.w;
  if (lane == 0) part -= t.x * t.x;
  float nsq = wred32(part);
  float xn  = fmaxf(sqrtf(nsq), MINNC);
  float cl  = fminf(xn, 15.0f);
  float s   = sinhf(cl) / xn;
  float4 o;
  o.y = s * t.y; o.z = s * t.z; o.w = s * t.w;
  o.x = (lane == 0) ? sqrtf(fmaxf(1.0f + s * s * nsq, EPSC)) : s * t.x;  // proj()
  row[lane] = o;
}

// ---------- per-edge: msg = logmap(out[src], x[dst]); atomic scatter ----------
// Both gathered node buffers (25.6MB each) are L2-resident on the 192MB L2,
// so this phase is L2-bandwidth bound, not HBM-bound.

__global__ void edge_kernel(const int* __restrict__ ei, const float* __restrict__ OUT,
                            const float* __restrict__ XL, float* __restrict__ AGG, int E) {
  int e    = blockIdx.x * (blockDim.x >> 5) + (threadIdx.x >> 5);
  int lane = threadIdx.x & 31;
  if (e >= E) return;
  int src = ei[e];
  int dst = ei[E + e];
  float4 o = reinterpret_cast<const float4*>(OUT + (size_t)src * 128)[lane];
  float4 y = reinterpret_cast<const float4*>(XL  + (size_t)dst * 128)[lane];
  float p1 = o.x * y.x + o.y * y.y + o.z * y.z + o.w * y.w;
  if (lane == 0) p1 -= 2.0f * o.x * y.x;
  float md = wred32(p1);                           // mdot(out, x_dst)
  float xy = fminf(md + 1.0f, -EPSC) - 1.0f;
  float4 u;
  u.x = y.x + xy * o.x; u.y = y.y + xy * o.y; u.z = y.z + xy * o.z; u.w = y.w + xy * o.w;
  float p2 = u.x * u.x + u.y * u.y + u.z * u.z + u.w * u.w;
  if (lane == 0) p2 -= 2.0f * u.x * u.x;
  float uu = wred32(p2);
  float nu = fmaxf(sqrtf(fmaxf(uu, EPSC)), MINNC);
  float ac = arcosh_f(fmaxf(-md, 1.0f + EPSC));
  float sc = sqrtf(fminf(ac * ac, 50.0f)) / nu;    // dist / nu
  float4 v;
  v.x = sc * u.x; v.y = sc * u.y; v.z = sc * u.z; v.w = sc * u.w;
  float p3 = o.x * v.x + o.y * v.y + o.z * v.z + o.w * v.w;
  if (lane == 0) p3 -= o.x * v.x;
  float ux = wred32(p3);                           // proj_tan numerator
  float o0 = __shfl(o.x, 0, 32);
  if (lane == 0) v.x = ux / fmaxf(o0, EPSC);
  float* ag = AGG + (size_t)dst * 128 + lane * 4;
  atomicAdd(ag + 0, v.x); atomicAdd(ag + 1, v.y);
  atomicAdd(ag + 2, v.z); atomicAdd(ag + 3, v.w);
}

// ---------- per-node tail: expmap -> poincare relu -> hyperboloid ----------

__global__ void node_tail_kernel(const float* __restrict__ AGG, const float* __restrict__ XL,
                                 float* __restrict__ XN, unsigned short* __restrict__ HC,
                                 int colbase, int N) {
  int node = blockIdx.x * (blockDim.x >> 5) + (threadIdx.x >> 5);
  int lane = threadIdx.x & 31;
  if (node >= N) return;
  float4 u  = reinterpret_cast<const float4*>(AGG + (size_t)node * 128)[lane];
  float4 xv = reinterpret_cast<const float4*>(XL  + (size_t)node * 128)[lane];
  float p1 = u.x * u.x + u.y * u.y + u.z * u.z + u.w * u.w;
  if (lane == 0) p1 -= 2.0f * u.x * u.x;
  float uu    = wred32(p1);
  float theta = fmaxf(fminf(sqrtf(fmaxf(uu, EPSC)), MAXNC), MINNC);
  float cl    = fminf(theta, 15.0f);
  float c = coshf(cl), s = sinhf(cl) / theta;
  float4 h;
  h.x = c * xv.x + s * u.x; h.y = c * xv.y + s * u.y;
  h.z = c * xv.z + s * u.z; h.w = c * xv.w + s * u.w;
  float p2 = h.x * h.x + h.y * h.y + h.z * h.z + h.w * h.w;
  if (lane == 0) p2 -= h.x * h.x;
  float hs = wred32(p2);
  float h0 = sqrtf(fmaxf(1.0f + hs, EPSC));
  float inv = 1.0f / (h0 + 1.0f);
  float4 p;
  p.x = fmaxf(h.x * inv, 0.0f); p.y = fmaxf(h.y * inv, 0.0f);
  p.z = fmaxf(h.z * inv, 0.0f); p.w = fmaxf(h.w * inv, 0.0f);
  if (lane == 0) p.x = 0.0f;
  float sq = wred32(p.x * p.x + p.y * p.y + p.z * p.z + p.w * p.w);
  float di = 1.0f / (1.0f - sq);
  float4 xo;
  xo.x = 2.0f * p.x * di; xo.y = 2.0f * p.y * di;
  xo.z = 2.0f * p.z * di; xo.w = 2.0f * p.w * di;
  if (lane == 0) xo.x = (1.0f + sq) * di;
  reinterpret_cast<float4*>(XN + (size_t)node * 128)[lane] = xo;
  union { unsigned short us[4]; uint2 v; } pk;
  pk.us[0] = f2bf(xo.x); pk.us[1] = f2bf(xo.y); pk.us[2] = f2bf(xo.z); pk.us[3] = f2bf(xo.w);
  *reinterpret_cast<uint2*>(HC + (size_t)node * 384 + colbase + lane * 4) = pk.v;
}

// ---------- bf16 WMMA GEMM: C[M,128] = A[M,K] * B[128,K]^T (+bias) ----------
// One wave owns a 16-row M-block with 8 accumulators (full 128-wide output),
// so A is loaded once per k-step. B (whole weight matrix) is staged into LDS
// once per workgroup by the Tensor Data Mover (TDM), using D#.pad to give LDS
// rows a 272B stride (bank-conflict-free ds_load_b128). Fallback: cooperative
// copy writing the identical padded layout.

__global__ void gemm_bf16_wmma(const unsigned short* __restrict__ A,
                               const unsigned short* __restrict__ Bg,
                               float* __restrict__ C,
                               const float* __restrict__ bias,
                               int M, int K) {
  extern __shared__ unsigned short Bsh[];
  const int nslab = K >> 7;

#if defined(HAVE_TDM)
  if (threadIdx.x < 32) {                       // one wave issues the DMA
    unsigned int ldsBase = (unsigned int)(size_t)(void*)&Bsh[0];
    for (int s = 0; s < nslab; ++s) {
      unsigned long long ga = (unsigned long long)(size_t)(Bg) + (unsigned long long)s * 256ull;
      u32x4 g0;
      g0[0] = 1u;                                               // count=1, user mode
      g0[1] = ldsBase + (unsigned int)(s * BSLAB * 2);          // lds_addr (bytes)
      g0[2] = (unsigned int)(ga & 0xFFFFFFFFull);               // global_addr[31:0]
      g0[3] = (unsigned int)((ga >> 32) & 0x01FFFFFFull) | (2u << 30); // addr[56:32] | type=2
      i32x8 g1;
      g1[0] = (2 << 16) | (1 << 20) | (5 << 22) | (3 << 25);    // data=4B, pad on, ivl=64dw, amt=4dw
      g1[1] = 64 << 16;                                         // tensor_dim0 = 64 dwords (slab)
      g1[2] = 128 << 16;                                        // tensor_dim1 = 128 rows
      g1[3] = 64 << 16;                                         // tile_dim0 = 64 dwords
      g1[4] = 128;                                              // tile_dim1 = 128 rows
      g1[5] = K >> 1;                                           // tensor_dim0_stride (dwords)
      g1[6] = 0;
      g1[7] = 0;
      i32x4 z4 = {0, 0, 0, 0};
#if __clang_major__ >= 23
      i32x8 z8 = {0, 0, 0, 0, 0, 0, 0, 0};
      __builtin_amdgcn_tensor_load_to_lds(g0, g1, z4, z4, z8, 0);
#else
      __builtin_amdgcn_tensor_load_to_lds(g0, g1, z4, z4, 0);
#endif
    }
    __builtin_amdgcn_s_wait_tensorcnt(0);
  }
#else
  {  // cooperative fallback copy into the same padded slab layout
    int chunks = nslab * 128 * 16;              // 8-ushort chunks
    for (int id = threadIdx.x; id < chunks; id += blockDim.x) {
      int s = id / (128 * 16), rem = id - s * (128 * 16);
      int r = rem >> 4, cc = rem & 15;
      uint4 d = *reinterpret_cast<const uint4*>(Bg + (size_t)r * K + s * 128 + cc * 8);
      *reinterpret_cast<uint4*>(&Bsh[(size_t)s * BSLAB + r * BROW + cc * 8]) = d;
    }
  }
#endif
  __syncthreads();

  int wave  = blockIdx.x * (blockDim.x >> 5) + (threadIdx.x >> 5);
  int lane  = threadIdx.x & 31;
  int mt    = (M + 15) >> 4;
  int tmRaw = wave << 4;
  int tm    = (wave < mt) ? tmRaw : (mt - 1) << 4;   // clamp; keep wave alive (EXEC all-1s)
  int nIdx  = lane & 15;
  int kh    = lane >> 4;
  int rowA  = tm + nIdx; if (rowA >= M) rowA = M - 1;
  const unsigned short* pa = A + (size_t)rowA * K + kh * 8;

  v8f acc[8];
#pragma unroll
  for (int t = 0; t < 8; ++t) acc[t] = (v8f){};

  for (int kb = 0; kb < K; kb += 32) {
    v8bf a0 = *reinterpret_cast<const v8bf*>(pa + kb);
    v8bf a1 = *reinterpret_cast<const v8bf*>(pa + kb + 16);
    v16bf av = __builtin_shufflevector(a0, a1, 0,1,2,3,4,5,6,7,8,9,10,11,12,13,14,15);
    const unsigned short* bbase = Bsh + (size_t)(kb >> 7) * BSLAB + (kb & 127) + kh * 8;
#pragma unroll
    for (int t = 0; t < 8; ++t) {
      const unsigned short* pb = bbase + (t * 16 + nIdx) * BROW;
      v8bf b0 = *reinterpret_cast<const v8bf*>(pb);
      v8bf b1 = *reinterpret_cast<const v8bf*>(pb + 16);
      v16bf bv = __builtin_shufflevector(b0, b1, 0,1,2,3,4,5,6,7,8,9,10,11,12,13,14,15);
      acc[t] = __builtin_amdgcn_wmma_f32_16x16x32_bf16(false, av, false, bv, (short)0, acc[t],
                                                       false, false);
    }
  }

  bool storeOK = (tmRaw < M);
  int rb = tm + kh * 8;                          // C layout: VGPR r -> M = khalf*8 + r
#pragma unroll
  for (int t = 0; t < 8; ++t) {
    int col = t * 16 + nIdx;
    float ba = bias ? bias[col] : 0.0f;
#pragma unroll
    for (int r = 0; r < 8; ++r) {
      int row = rb + r;
      if (storeOK && row < M) C[(size_t)row * 128 + col] = acc[t][r] + ba;
    }
  }
}

// ---------- orchestration ----------

extern "C" void kernel_launch(void* const* d_in, const int* in_sizes, int n_in,
                              void* d_out, int out_size, void* d_ws, size_t ws_size,
                              hipStream_t stream) {
  const float* x  = (const float*)d_in[0];
  const int*   ei = (const int*)d_in[1];
  const float* W0 = (const float*)d_in[2];
  const float* W1 = (const float*)d_in[3];
  const float* Wf = (const float*)d_in[4];
  const float* bf = (const float*)d_in[5];
  float* out = (float*)d_out;

  const int N  = in_sizes[0] / 128;
  const int E  = in_sizes[1] / 2;
  const int Kf = in_sizes[4] / 128;            // 384

  size_t n128 = (size_t)N * 128;
  float* A1 = (float*)d_ws;                    // GEMM out -> expmap0 out
  float* A2 = A1 + n128;                       // aggr
  float* A3 = A2 + n128;                       // x of current layer
  unsigned short* B1  = (unsigned short*)(A3 + n128);   // logmap0 bf16 (N x 128)
  unsigned short* HC  = B1 + n128;                      // concat bf16 (N x 384)
  unsigned short* Wb  = HC + (size_t)N * 384;           // conv weight bf16
  unsigned short* Wfb = Wb + 128 * 128;                 // final weight bf16

  const int TB = 256;                          // 8 waves / block
  int nodeBlocks = (N + 7) / 8;
  int edgeBlocks = (E + 7) / 8;
  int elemBlocks = (N * 128 + TB - 1) / TB;
  int mt         = (N + 15) / 16;
  int gemmBlocks = (mt + 7) / 8;

  x_to_hcat_kernel<<<elemBlocks, TB, 0, stream>>>(x, HC, N);

  const float* xin = x;
  for (int layer = 0; layer < 2; ++layer) {
    const float* W = (layer == 0) ? W0 : W1;
    f2bf_kernel<<<(128 * 128 + TB - 1) / TB, TB, 0, stream>>>(W, Wb, 128 * 128);
    logmap0_bf16_kernel<<<nodeBlocks, TB, 0, stream>>>(xin, B1, N);
    gemm_bf16_wmma<<<gemmBlocks, TB, (size_t)(128 / 128) * BSLAB * 2, stream>>>(
        B1, Wb, A1, nullptr, N, 128);
    expmap0_kernel<<<nodeBlocks, TB, 0, stream>>>(A1, N);
    zero_f32_kernel<<<2048, TB, 0, stream>>>(A2, n128);
    edge_kernel<<<edgeBlocks, TB, 0, stream>>>(ei, A1, xin, A2, E);
    node_tail_kernel<<<nodeBlocks, TB, 0, stream>>>(A2, xin, A3, HC, 128 * (layer + 1), N);
    xin = A3;
  }

  f2bf_kernel<<<(128 * Kf + TB - 1) / TB, TB, 0, stream>>>(Wf, Wfb, 128 * Kf);
  gemm_bf16_wmma<<<gemmBlocks, TB, (size_t)(Kf / 128) * BSLAB * 2, stream>>>(
      HC, Wfb, out, bf, N, Kf);
}
